// PPM_11244224381572
// MI455X (gfx1250) — matrix operand order repaired and checked by reading, not compiled, not run on hardware
//
#include <hip/hip_runtime.h>

// ---------------------------------------------------------------------------
// Problem constants (from reference): x [4,256,64,64], W [256,256], b [256]
// ---------------------------------------------------------------------------
#define B_DIM 4
#define C_DIM 256
#define P_DIM 4096            // 64*64 pixels
#define EPS_NORM 1e-8f

typedef __attribute__((ext_vector_type(16))) __bf16 v16bf;
typedef __attribute__((ext_vector_type(8)))  float  v8f;

static __device__ __forceinline__ v8f wmma_bf16(v16bf a, v16bf b, v8f c) {
  // D(f32 16x16) = A(16x32 bf16) * B(32x16 bf16) + C
  return __builtin_amdgcn_wmma_f32_16x16x32_bf16(
      /*neg_a=*/false, a, /*neg_b=*/false, b,
      /*c_mod=*/(short)0, c, /*reuse_a=*/false, /*reuse_b=*/false);
}

// ---------------------------------------------------------------------------
// Kernel 1: rnorm[b,p] = 1 / max(||x[b,:,p]||, eps)
// ---------------------------------------------------------------------------
__global__ void ppm_norm_kernel(const float* __restrict__ x,
                                float* __restrict__ rnorm) {
  int e = blockIdx.x * 256 + threadIdx.x;          // < B*P = 16384
  int b = e >> 12, p = e & (P_DIM - 1);
  const float* xp = x + ((size_t)b << 20) + p;     // b*C*P
  float s = 0.f;
  #pragma unroll 8
  for (int c = 0; c < C_DIM; ++c) {
    float v = xp[(size_t)c << 12];
    s += v * v;
  }
  rnorm[e] = 1.0f / fmaxf(sqrtf(s), EPS_NORM);
}

// ---------------------------------------------------------------------------
// Kernel 2: pack normalized x into bf16 WMMA tiles (both operand layouts).
//   xnA: A-layout 16x32 (M=pixel, K=c) per (b, tile16, cktile32): 512 bf16
//        lane = m + 16*((k%16)/8), slot = (k/16)*8 + (k%8)
//   xnB: B-layout 32x16 (K=c, N=pixel) per (b, tile16, cktile32): 512 bf16
//        lane = n + 16*((k%16)/8), slot = (k/16)*8 + (k%8)
// ---------------------------------------------------------------------------
__global__ void ppm_pack_xn_kernel(const float* __restrict__ x,
                                   const float* __restrict__ rnorm,
                                   __bf16* __restrict__ xnA,
                                   __bf16* __restrict__ xnB) {
  int e = blockIdx.x * 256 + threadIdx.x;          // < 4*256*8*512 = 4194304
  int r    = e & 511;
  int tile = e >> 9;
  int lane = r >> 4, slot = r & 15;
  int ck  = tile & 7;
  int t16 = (tile >> 3) & 255;
  int b   = tile >> 11;

  {  // A-layout element
    int m  = lane & 15, hi = lane >> 4;
    int k  = ((slot >> 3) << 4) + hi * 8 + (slot & 7);
    int c  = ck * 32 + k;
    int p  = t16 * 16 + m;
    float v = x[(((size_t)b * C_DIM + c) << 12) + p] * rnorm[(b << 12) + p];
    xnA[e] = (__bf16)v;
  }
  {  // B-layout element
    int n  = lane & 15;
    int k  = ((slot >> 3) << 4) + (lane >> 4) * 8 + (slot & 7);
    int c  = ck * 32 + k;
    int q  = t16 * 16 + n;
    float v = x[(((size_t)b * C_DIM + c) << 12) + q] * rnorm[(b << 12) + q];
    xnB[e] = (__bf16)v;
  }
}

// ---------------------------------------------------------------------------
// Kernel 3: pack W (256x256 f32) into bf16 A-layout tiles (16o x 32c).
// Tiny one-time cost: 65536 elements = 128 KB.
// ---------------------------------------------------------------------------
__global__ void ppm_pack_w_kernel(const float* __restrict__ W,
                                  __bf16* __restrict__ WA) {
  int e = blockIdx.x * 256 + threadIdx.x;          // < 16*8*512 = 65536
  int r = e & 511, tile = e >> 9;
  int ck = tile & 7, ot = tile >> 3;
  int lane = r >> 4, slot = r & 15;
  int m  = lane & 15, hi = lane >> 4;
  int k  = ((slot >> 3) << 4) + hi * 8 + (slot & 7);
  int o  = ot * 16 + m;
  int c  = ck * 32 + k;
  WA[e] = (__bf16)W[o * C_DIM + c];
}

// ---------------------------------------------------------------------------
// Kernel 4: gx = W @ x + bias via WMMA, using gx = (W @ xn) * max(||x||,eps) + bias.
// One wave per (b, q-tile16): B operand = resident xnB tiles, A = streamed W
// tiles. Output written directly as bf16 B-layout tiles (K=q32, N=c16) for the
// second GEMM of the main kernel.
// ---------------------------------------------------------------------------
__global__ void __launch_bounds__(256)
ppm_gx_kernel(const __bf16* __restrict__ WA,
              const __bf16* __restrict__ xnB,
              const float* __restrict__ rnorm,
              const float* __restrict__ bias,
              __bf16* __restrict__ gxB) {
  const int wave = threadIdx.x >> 5;
  const int lane = threadIdx.x & 31;
  const int hi   = lane >> 4;
  const int nn   = lane & 15;

  const int qtg = blockIdx.x * 8 + wave;           // global q-tile16, < B*256
  const int b   = qtg >> 8;
  const int qt  = qtg & 255;

  // resident B operand: xn for this q-tile, all 8 K-chunks
  v16bf bq[8];
  #pragma unroll
  for (int ck = 0; ck < 8; ++ck)
    bq[ck] = *(const v16bf*)(xnB + ((((size_t)b * 256 + qt) * 8 + ck) << 9) + lane * 16);

  // each lane owns one q column (C-layout n = lane%16): its un-normalizer
  const float nv = 1.0f / rnorm[(b << 12) + qt * 16 + nn];

  const int qk = qt >> 1;                          // 32-q block index
  const int lane_tgt = ((qt & 1) << 3);            // +8 slots for upper k half

  #pragma unroll 4
  for (int ot = 0; ot < 16; ++ot) {
    v8f acc = {};
    #pragma unroll
    for (int ck = 0; ck < 8; ++ck) {
      v16bf aw = *(const v16bf*)(WA + (((ot * 8 + ck) << 9)) + lane * 16);
      acc = wmma_bf16(aw, bq[ck], acc);
    }
    // bias for o = ot*16 + v + 8*hi, v=0..7 : 8 consecutive floats
    const float* bptr = bias + ot * 16 + 8 * hi;
    float4 bb0 = *(const float4*)bptr;
    float4 bb1 = *(const float4*)(bptr + 4);
    float bv[8] = {bb0.x, bb0.y, bb0.z, bb0.w, bb1.x, bb1.y, bb1.z, bb1.w};
    // scatter-store into gxB B-layout: (k=q&31, n=o&15)
    __bf16* gout = gxB + ((((size_t)b * 128 + qk) * 16 + ot) << 9);
    #pragma unroll
    for (int v = 0; v < 8; ++v) {
      float g = acc[v] * nv + bv[v];
      int lt = (v + 8 * hi) + ((nn >> 3) << 4);    // target lane in B-layout
      gout[lt * 16 + (nn & 7) + lane_tgt] = (__bf16)g;
    }
  }
}

// ---------------------------------------------------------------------------
// Kernel 5 (fused hot loop): one workgroup = (batch b, 16-pixel p-tile).
// 8 wave32s; wave w owns q in [w*512, (w+1)*512).
//
// Similarity GEMM is computed TRANSPOSED (M=q, N=p): its f32 C-layout output
// maps lane-for-lane onto the A-layout the second GEMM needs, so
// relu()^2 -> bf16 is pure in-lane VALU. No LDS, no barriers in the loop.
// Per 32-q block: 16 WMMAs (s-tile, K=256) + 16 WMMAs (apply to gx, N=256c).
// Cross-wave reduction of y[16x256] via LDS ds_add_f32 at the end.
// ---------------------------------------------------------------------------
__global__ void __launch_bounds__(256)
ppm_main_kernel(const __bf16* __restrict__ xnA,
                const __bf16* __restrict__ xnB,
                const __bf16* __restrict__ gxB,
                float* __restrict__ out) {
  extern __shared__ char smem_raw[];
  float* yred = (float*)smem_raw;                  // 16*256 f32 = 16 KB

  const int tid  = threadIdx.x;
  const int wave = tid >> 5;
  const int lane = tid & 31;
  const int hi   = lane >> 4;
  const int nn   = lane & 15;
  const int b    = blockIdx.x >> 8;                // 256 p-tiles per batch
  const int pt   = blockIdx.x & 255;

  // zero the reduction buffer
  #pragma unroll
  for (int i = 0; i < 16; ++i) yred[tid + 256 * i] = 0.0f;
  __syncthreads();

  // resident B operand: this p-tile's xn, all 8 K-chunks (K=256 channels)
  v16bf bp[8];
  #pragma unroll
  for (int ck = 0; ck < 8; ++ck)
    bp[ck] = *(const v16bf*)(xnB + ((((size_t)b * 256 + pt) * 8 + ck) << 9) + lane * 16);

  v8f yacc[16] = {};                               // 16 x (16x16 f32) = all 256 channels

  for (int qb = 0; qb < 16; ++qb) {
    const int qk = wave * 16 + qb;                 // 32-pixel q block index (0..127)

    const __bf16* abase0 = xnA + (((size_t)(b * 256 + 2 * qk) * 8) << 9) + lane * 16;
    const __bf16* abase1 = abase0 + 8 * 512;
    const __bf16* gbase  = gxB + ((((size_t)b * 128 + qk) * 16) << 9) + lane * 16;

    if (qb + 1 < 16) {                             // prefetch next q block
      __builtin_prefetch(abase0 + 16 * 512, 0, 1);
      __builtin_prefetch(gbase  + 16 * 512, 0, 1);
    }

    // ---- transposed similarity tile: D[q x p] = xn_q(16x256) x xn_p(256x16)
    v8f s0 = {}, s1 = {};
    #pragma unroll
    for (int ck = 0; ck < 8; ++ck) {
      v16bf a0 = *(const v16bf*)(abase0 + ck * 512);
      v16bf a1 = *(const v16bf*)(abase1 + ck * 512);
      s0 = wmma_bf16(a0, bp[ck], s0);
      s1 = wmma_bf16(a1, bp[ck], s1);
    }

    // ---- relu^2 + bf16: C-layout of D[q][p] == A-layout of s[p][q] per lane:
    //   slot j   (j<8):  k_q = 8*hi + j        == s0 reg j
    //   slot 8+j (j<8):  k_q = 16 + 8*hi + j   == s1 reg j
    v16bf sA;
    #pragma unroll
    for (int j = 0; j < 8; ++j) {
      float t0 = fmaxf(s0[j], 0.0f); t0 *= t0;
      float t1 = fmaxf(s1[j], 0.0f); t1 *= t1;
      sA[j]     = (__bf16)t0;
      sA[8 + j] = (__bf16)t1;
    }

    // ---- y[16p x 256c] += s(16x32) x gx_B(32x16) per channel tile ----
    #pragma unroll
    for (int ct = 0; ct < 16; ++ct) {
      v16bf bg = *(const v16bf*)(gbase + ct * 512);
      yacc[ct] = wmma_bf16(sA, bg, yacc[ct]);
    }
  }

  // ---- cross-wave reduction via LDS float atomics (ds_add_f32) ----
  #pragma unroll
  for (int ct = 0; ct < 16; ++ct) {
    #pragma unroll
    for (int v = 0; v < 8; ++v) {
      int m = v + 8 * hi;
      atomicAdd(&yred[m * 256 + ct * 16 + nn], yacc[ct][v]);
    }
  }
  __syncthreads();

  // ---- store: thread tid owns channel c=tid, 16 consecutive pixels ----
  {
    const int c = tid;
    float* orow = out + (((size_t)b * C_DIM + c) << 12) + pt * 16;
    #pragma unroll
    for (int m4 = 0; m4 < 4; ++m4) {
      float4 v;
      v.x = yred[(m4 * 4 + 0) * 256 + c];
      v.y = yred[(m4 * 4 + 1) * 256 + c];
      v.z = yred[(m4 * 4 + 2) * 256 + c];
      v.w = yred[(m4 * 4 + 3) * 256 + c];
      *(float4*)(orow + m4 * 4) = v;
    }
  }
}

// ---------------------------------------------------------------------------
// Host launcher
// ---------------------------------------------------------------------------
extern "C" void kernel_launch(void* const* d_in, const int* in_sizes, int n_in,
                              void* d_out, int out_size, void* d_ws, size_t ws_size,
                              hipStream_t stream) {
  const float* x    = (const float*)d_in[0];   // [4,256,64,64]
  const float* W    = (const float*)d_in[1];   // [256,256]
  const float* bias = (const float*)d_in[2];   // [256]
  float* out = (float*)d_out;

  // workspace partition
  float*  rnorm = (float*)d_ws;                              // 16384 f32   (64 KB)
  __bf16* xnA   = (__bf16*)((char*)d_ws + 65536);            // 4194304 bf16 (8 MB)
  __bf16* xnB   = xnA + 4194304;                             // 8 MB
  __bf16* gxB   = xnB + 4194304;                             // 8 MB
  __bf16* WA    = gxB + 4194304;                             // 65536 bf16 (128 KB)

  ppm_norm_kernel<<<64, 256, 0, stream>>>(x, rnorm);
  ppm_pack_xn_kernel<<<16384, 256, 0, stream>>>(x, rnorm, xnA, xnB);
  ppm_pack_w_kernel<<<256, 256, 0, stream>>>(W, WA);
  ppm_gx_kernel<<<128, 256, 0, stream>>>(WA, xnB, rnorm, bias, gxB);
  ppm_main_kernel<<<B_DIM * 256, 256, 16384, stream>>>(xnA, xnB, gxB, out);
}
